// Network_40596030882208
// MI455X (gfx1250) — matrix operand order repaired and checked by reading, not compiled, not run on hardware
//
#include <hip/hip_runtime.h>

// ---------------- types / constants ----------------
typedef __attribute__((ext_vector_type(16))) __bf16 v16bf;
typedef __attribute__((ext_vector_type(8)))  float  v8f;
typedef __attribute__((ext_vector_type(4)))  unsigned int v4u;
typedef __attribute__((ext_vector_type(8)))  int v8i_t;
typedef __attribute__((ext_vector_type(4)))  int v4i_t;

#define B_   4
#define C_   64
#define H_   128
#define W_   128
#define HW_  16384

__device__ __forceinline__ unsigned short f2bf(float f) {
  union { float f; unsigned u; } v; v.f = f;
  unsigned r = (v.u + 0x7FFFu + ((v.u >> 16) & 1u)) >> 16;   // RNE to bf16
  return (unsigned short)r;
}
__device__ __forceinline__ __bf16 bfbits(unsigned short s) {
  union { unsigned short u; __bf16 b; } v; v.u = s; return v.b;
}

// ---------------- weight packing (fp32 -> bf16, WMMA-A friendly) ----------------
// out[(t*Co + co)*Ci + ci] = w[(co*Ci + ci)*KK + t]
__global__ void k_pack_conv(const float* __restrict__ w, unsigned short* __restrict__ out,
                            int Co, int Ci, int KK) {
  int i = blockIdx.x * 256 + threadIdx.x;
  int total = Co * Ci * KK;
  if (i >= total) return;
  int t  = i / (Co * Ci);
  int r  = i % (Co * Ci);
  int co = r / Ci, ci = r % Ci;
  out[i] = f2bf(w[(co * Ci + ci) * KK + t]);
}
// Wcp[(r*16+k)*512 + c] = w_cp1[(r*512 + c)*16 + k]   ([16,512,16])
__global__ void k_pack_cp(const float* __restrict__ w, unsigned short* __restrict__ out) {
  int i = blockIdx.x * 256 + threadIdx.x;
  if (i >= 256 * 512) return;
  int row = i >> 9, c = i & 511;
  int r = row >> 4, k = row & 15;
  out[i] = f2bf(w[(r * 512 + c) * 16 + k]);
}
// W2[o*256 + r*16 + p] = w_out3[(p*64 + o)*16 + r]    ([16,64,16])
__global__ void k_pack_w2(const float* __restrict__ w, unsigned short* __restrict__ out) {
  int i = blockIdx.x * 256 + threadIdx.x;
  if (i >= 64 * 256) return;
  int o = i >> 8, rp = i & 255;
  int r = rp >> 4, p = rp & 15;
  out[i] = f2bf(w[(p * 64 + o) * 16 + r]);
}

// ---------------- generic 1x1-GEMM over pixels, WMMA bf16 ----------------
// Y[b, out_coff+m, pix] = sum_ci Wp[m][ci] * X[b, ci, pix]
// A fragment: lane m=lane&15, elem e   -> K=16h+e
// B fragment: lane n=lane&15, elem e   -> K=16h+e
__global__ void k_gemm_wmma(const unsigned short* __restrict__ Wp,
                            const float* __restrict__ Xf, const unsigned short* __restrict__ Xb,
                            float* __restrict__ Yf, unsigned short* __restrict__ Yb,
                            const float* __restrict__ scale_tab, int scale_idx,
                            int Ci, int Co, int relu_in, int out_cstride, int out_coff) {
  int lane = threadIdx.x & 31, wv = threadIdx.x >> 5;
  int tile = blockIdx.x * 8 + wv;
  int cot  = Co >> 4;
  int co_t = tile % cot, px_t = tile / cot;
  int mn   = lane & 15, hh = lane >> 4;
  int pix  = px_t * 16 + mn;
  int b    = blockIdx.z;

  const __bf16* Arow = ((const __bf16*)Wp) + (size_t)(co_t * 16 + mn) * Ci;
  v8f acc = {};
  for (int kc = 0; kc < Ci; kc += 32) {
    v16bf a, bb;
    const __bf16* ar = Arow + kc + 16 * hh;
#pragma unroll
    for (int j = 0; j < 8; ++j) a[j] = ar[j];
#pragma unroll
    for (int j = 0; j < 8; ++j) a[8 + j] = ar[8 + j];
    size_t cbase = ((size_t)b * Ci + kc + 16 * hh) * HW_ + pix;
    if (Xb) {
      const __bf16* xp = (const __bf16*)Xb + cbase;
#pragma unroll
      for (int j = 0; j < 16; ++j) bb[j] = xp[(size_t)j * HW_];
    } else {
      const float* xp = Xf + cbase;
#pragma unroll
      for (int j = 0; j < 16; ++j) {
        float v = xp[(size_t)j * HW_];
        if (relu_in) v = fmaxf(v, 0.f);
        bb[j] = bfbits(f2bf(v));
      }
    }
    acc = __builtin_amdgcn_wmma_f32_16x16x32_bf16(false, a, false, bb, (short)0, acc,
                                                  false, false);
  }
  float scl = scale_tab ? scale_tab[b * 32 + scale_idx] : 1.0f;
#pragma unroll
  for (int i = 0; i < 8; ++i) {
    int m = co_t * 16 + i + 8 * hh;
    size_t o = ((size_t)b * out_cstride + out_coff + m) * HW_ + pix;
    float v = acc[i] * scl;
    if (Yb) Yb[o] = f2bf(v);
    else    Yf[o] = v;
  }
}

// ---------------- CP contraction with TDM-staged B tile ----------------
// out1[b, m, pix] = sum_{c<512} Wcp[m][c] * states[b, c, pix]
// One tensor_load_to_lds DMA per block stages the [512 ch x 16 px] bf16 tile
// (global row stride HW_) into LDS; 8 waves x grid.y=2 cover the 16 Co-tiles.
__global__ void __launch_bounds__(256) k_cp_gemm_tdm(const unsigned short* __restrict__ Wp,
                                                     const unsigned short* __restrict__ states,
                                                     float* __restrict__ out1) {
  __shared__ __align__(16) unsigned short tileB[512 * 16];   // [ch][px], 16 KiB
  int b   = blockIdx.z;
  int pxb = blockIdx.x * 16;
  int lane = threadIdx.x & 31, wv = threadIdx.x >> 5;

#if __has_builtin(__builtin_amdgcn_tensor_load_to_lds)
  if (threadIdx.x == 0) {   // wave 0 issues the DMA (other waves branch around, EXEC==0)
    unsigned long long ga =
        (unsigned long long)(uintptr_t)(states + (size_t)b * 512 * HW_ + pxb);
    unsigned lds = (unsigned)(uintptr_t)&tileB[0];
    v4u g0;
    g0[0] = 1u;                                        // count=1, user descriptor
    g0[1] = lds;                                       // lds_addr (bytes)
    g0[2] = (unsigned)(ga & 0xFFFFFFFFu);              // global_addr[31:0]
    g0[3] = (unsigned)((ga >> 32) & 0x01FFFFFFu)       // global_addr[56:32]
            | (2u << 30);                              // type = 2 ("image")
    v8i_t g1;
    g1[0] = (1 << 16);     // workgroup_mask=0, data_size=1 (2 bytes)
    g1[1] = (16 << 16);    // tensor_dim0[15:0] = 16
    g1[2] = (512 << 16);   // tensor_dim0[31:16]=0 | tensor_dim1[15:0] = 512
    g1[3] = (16 << 16);    // tensor_dim1[31:16]=0 | tile_dim0 = 16
    g1[4] = 512;           // tile_dim1 = 512 | tile_dim2 = 0
    g1[5] = HW_;           // tensor_dim0_stride[31:0] = 16384
    g1[6] = 0;             // stride[47:32]=0 | tensor_dim1_stride lo = 0
    g1[7] = 0;
    v4i_t z4 = {0, 0, 0, 0};
#if __has_include(<hip/amd_detail/amd_gfx1250_TDM.h>)
    v8i_t z8 = {0, 0, 0, 0, 0, 0, 0, 0};
    __builtin_amdgcn_tensor_load_to_lds(g0, g1, z4, z4, z8, 0);   // clang-23 form
#else
    __builtin_amdgcn_tensor_load_to_lds(g0, g1, z4, z4, 0);       // ROCm 7.2 form
#endif
    __builtin_amdgcn_s_wait_tensorcnt(0);
  }
#else
  for (int i = threadIdx.x; i < 512 * 16; i += 256) {
    int ch = i >> 4, px = i & 15;
    tileB[i] = states[((size_t)b * 512 + ch) * HW_ + pxb + px];
  }
#endif
  __syncthreads();

  int mn = lane & 15, hh = lane >> 4;
  int pix = pxb + mn;
  int co_t = blockIdx.y * 8 + wv;                       // 16 Co tiles
  const __bf16* Arow = ((const __bf16*)Wp) + (size_t)(co_t * 16 + mn) * 512;
  v8f acc = {};
  for (int kc = 0; kc < 512; kc += 32) {
    v16bf a, bb;
    const __bf16* ar = Arow + kc + 16 * hh;
#pragma unroll
    for (int j = 0; j < 8; ++j) a[j] = ar[j];
#pragma unroll
    for (int j = 0; j < 8; ++j) a[8 + j] = ar[8 + j];
    const unsigned short* tb = &tileB[(kc + 16 * hh) * 16 + mn];
#pragma unroll
    for (int j = 0; j < 16; ++j) bb[j] = bfbits(tb[j * 16]);
    acc = __builtin_amdgcn_wmma_f32_16x16x32_bf16(false, a, false, bb, (short)0, acc,
                                                  false, false);
  }
#pragma unroll
  for (int i = 0; i < 8; ++i) {
    int m = co_t * 16 + i + 8 * hh;
    out1[((size_t)b * 256 + m) * HW_ + pix] = acc[i];
  }
}

// ---------------- 3x3 conv 64->64 as 9-tap implicit GEMM, WMMA bf16 ----------------
__global__ void k_conv3x3_wmma(const float* __restrict__ X, const unsigned short* __restrict__ Wp,
                               float* __restrict__ Y, const float* __restrict__ Res, int relu_in) {
  int lane = threadIdx.x & 31, wv = threadIdx.x >> 5;
  int tile = blockIdx.x * 8 + wv;
  int co_t = tile & 3, px_t = tile >> 2;
  int mn   = lane & 15, hh = lane >> 4;
  int pix  = px_t * 16 + mn;
  int h0   = pix >> 7, w0 = pix & 127;
  int b    = blockIdx.z;
  v8f acc = {};
  for (int t = 0; t < 9; ++t) {
    int yy = h0 + t / 3 - 1, xx = w0 + t % 3 - 1;
    bool valid = ((unsigned)yy < (unsigned)H_) && ((unsigned)xx < (unsigned)W_);
    const float* xp0 = X + (size_t)b * C_ * HW_ + (ptrdiff_t)yy * W_ + xx;
    const __bf16* At = (const __bf16*)Wp + ((size_t)t * C_ + co_t * 16 + mn) * C_;
    for (int kc = 0; kc < 64; kc += 32) {
      v16bf a, bb;
      const __bf16* ar = At + kc + 16 * hh;
#pragma unroll
      for (int j = 0; j < 8; ++j) a[j] = ar[j];
#pragma unroll
      for (int j = 0; j < 8; ++j) a[8 + j] = ar[8 + j];
      const float* xp = xp0 + (size_t)(kc + 16 * hh) * HW_;
#pragma unroll
      for (int j = 0; j < 16; ++j) {
        float v = 0.f;
        if (valid) v = xp[(size_t)j * HW_];
        if (relu_in) v = fmaxf(v, 0.f);
        bb[j] = bfbits(f2bf(v));
      }
      acc = __builtin_amdgcn_wmma_f32_16x16x32_bf16(false, a, false, bb, (short)0, acc,
                                                    false, false);
    }
  }
#pragma unroll
  for (int i = 0; i < 8; ++i) {
    int m = co_t * 16 + i + 8 * hh;
    size_t o = ((size_t)b * C_ + m) * HW_ + pix;
    float v = acc[i];
    if (Res) v += Res[o];
    Y[o] = v;
  }
}

// ---------------- small direct convs (3->64, 64->3) ----------------
__global__ void k_conv_in(const float* __restrict__ x, const float* __restrict__ w,
                          float* __restrict__ Y) {
  int idx = blockIdx.x * 256 + threadIdx.x;           // B*64*HW
  int pix = idx & (HW_ - 1);
  int co  = (idx >> 14) & 63;
  int b   = idx >> 20;
  int h0 = pix >> 7, w0 = pix & 127;
  float acc = 0.f;
  for (int ci = 0; ci < 3; ++ci)
    for (int ky = 0; ky < 3; ++ky) {
      int yy = h0 + ky - 1; if ((unsigned)yy >= (unsigned)H_) continue;
      for (int kx = 0; kx < 3; ++kx) {
        int xx = w0 + kx - 1; if ((unsigned)xx >= (unsigned)W_) continue;
        acc += x[((size_t)b * 3 + ci) * HW_ + yy * W_ + xx] *
               w[((co * 3 + ci) * 3 + ky) * 3 + kx];
      }
    }
  Y[idx] = acc;
}
__global__ void k_conv_out(const float* __restrict__ X, const float* __restrict__ w,
                           float* __restrict__ out) {
  int idx = blockIdx.x * 256 + threadIdx.x;           // B*3*HW
  if (idx >= B_ * 3 * HW_) return;
  int pix = idx & (HW_ - 1);
  int rest = idx >> 14;
  int co = rest % 3, b = rest / 3;
  int h0 = pix >> 7, w0 = pix & 127;
  float acc = 0.f;
  for (int ci = 0; ci < 64; ++ci) {
    const float* xp = X + ((size_t)b * 64 + ci) * HW_;
    const float* wp = w + (size_t)(co * 64 + ci) * 9;
    for (int ky = 0; ky < 3; ++ky) {
      int yy = h0 + ky - 1; if ((unsigned)yy >= (unsigned)H_) continue;
      for (int kx = 0; kx < 3; ++kx) {
        int xx = w0 + kx - 1; if ((unsigned)xx >= (unsigned)W_) continue;
        acc += xp[yy * W_ + xx] * wp[ky * 3 + kx];
      }
    }
  }
  out[idx] = acc;
}

// ---------------- depthwise conv (SAME, dilated) on pre-relu'd input ----------------
__global__ void k_dwconv(const float* __restrict__ X, const float* __restrict__ Wd,
                         float* __restrict__ Y, int K, int dil) {
  int idx = blockIdx.x * 256 + threadIdx.x;           // B*64*HW
  int pix = idx & (HW_ - 1);
  int c   = (idx >> 14) & 63;
  int h0 = pix >> 7, w0 = pix & 127;
  int pad = dil * (K - 1) / 2;
  const float* xc = X + (size_t)(idx - pix);
  const float* wc = Wd + c * K * K;
  float acc = 0.f;
  for (int ky = 0; ky < K; ++ky) {
    int yy = h0 + ky * dil - pad; if ((unsigned)yy >= (unsigned)H_) continue;
    for (int kx = 0; kx < K; ++kx) {
      int xx = w0 + kx * dil - pad; if ((unsigned)xx >= (unsigned)W_) continue;
      acc += xc[yy * W_ + xx] * wc[ky * K + kx];
    }
  }
  Y[idx] = acc;
}

// ---------------- avg pool 3x3 /9, scaled, into states slot 7 (bf16) ----------------
__global__ void k_avgpool_states(const float* __restrict__ X, unsigned short* __restrict__ states,
                                 const float* __restrict__ attn, int widx) {
  int idx = blockIdx.x * 256 + threadIdx.x;
  int pix = idx & (HW_ - 1);
  int c   = (idx >> 14) & 63;
  int b   = idx >> 20;
  int h0 = pix >> 7, w0 = pix & 127;
  const float* xc = X + (size_t)(idx - pix);
  float s = 0.f;
  for (int ky = -1; ky <= 1; ++ky) {
    int yy = h0 + ky; if ((unsigned)yy >= (unsigned)H_) continue;
    for (int kx = -1; kx <= 1; ++kx) {
      int xx = w0 + kx; if ((unsigned)xx >= (unsigned)W_) continue;
      s += xc[yy * W_ + xx];
    }
  }
  float v = s * (1.f / 9.f) * attn[b * 32 + widx];
  states[((size_t)b * 512 + 448 + c) * HW_ + pix] = f2bf(v);
}

// ---------------- relu ----------------
__global__ void k_relu(const float* __restrict__ X, float* __restrict__ Y) {
  int idx = blockIdx.x * 256 + threadIdx.x;
  Y[idx] = fmaxf(X[idx], 0.f);
}

// ---------------- per-pixel M2 = M*M (16x16), LDS-tiled ----------------
__global__ void k_m2(const float* __restrict__ out1, float* __restrict__ M2) {
  __shared__ float sm[16 * 257];
  int b  = blockIdx.y;
  int pb = blockIdx.x * 16;
  int t  = threadIdx.x;
  const size_t base = (size_t)b * 256 * HW_;
  for (int i = t; i < 4096; i += 256) {
    int ch = i >> 4, px = i & 15;
    sm[px * 257 + ch] = out1[base + (size_t)ch * HW_ + pb + px];
  }
  __syncthreads();
  int px = t & 15, r = t >> 4;
  const float* Mp = &sm[px * 257];
  for (int p = 0; p < 16; ++p) {
    float acc = 0.f;
#pragma unroll
    for (int k = 0; k < 16; ++k) acc += Mp[r * 16 + k] * Mp[k * 16 + p];
    M2[base + (size_t)(r * 16 + p) * HW_ + pb + px] = acc;
  }
}

// ---------------- instance-norm + softsign + residual + relu ----------------
__global__ void k_norm(const float* __restrict__ Yin, const float* __restrict__ Res,
                       float* __restrict__ Sout) {
  __shared__ float s1[256], s2[256];
  __shared__ float stats[2];
  size_t base = ((size_t)blockIdx.y * 64 + blockIdx.x) * HW_;
  int t = threadIdx.x;
  float s = 0.f, q = 0.f;
  for (int i = t; i < HW_; i += 256) { float v = Yin[base + i]; s += v; q += v * v; }
  s1[t] = s; s2[t] = q; __syncthreads();
  for (int off = 128; off; off >>= 1) {
    if (t < off) { s1[t] += s1[t + off]; s2[t] += s2[t + off]; }
    __syncthreads();
  }
  if (t == 0) {
    float m = s1[0] * (1.f / HW_);
    float var = s2[0] * (1.f / HW_) - m * m;
    stats[0] = m; stats[1] = rsqrtf(var + 1e-5f);
  }
  __syncthreads();
  float m = stats[0], iv = stats[1];
  for (int i = t; i < HW_; i += 256) {
    float z = (Yin[base + i] - m) * iv;
    z = z / (1.f + fabsf(z));
    z += Res[base + i];
    Sout[base + i] = fmaxf(z, 0.f);
  }
}

// ---------------- global average pool ----------------
__global__ void k_gap(const float* __restrict__ X, float* __restrict__ pooled) {
  __shared__ float s1[256];
  size_t base = ((size_t)blockIdx.y * 64 + blockIdx.x) * HW_;
  int t = threadIdx.x;
  float s = 0.f;
  for (int i = t; i < HW_; i += 256) s += X[base + i];
  s1[t] = s; __syncthreads();
  for (int off = 128; off; off >>= 1) {
    if (t < off) s1[t] += s1[t + off];
    __syncthreads();
  }
  if (t == 0) pooled[blockIdx.y * 64 + blockIdx.x] = s1[0] * (1.f / HW_);
}

// ---------------- attention MLP + softmax ----------------
__global__ void k_oa(const float* __restrict__ pooled,
                     const float* __restrict__ w1, const float* __restrict__ b1,
                     const float* __restrict__ w2, const float* __restrict__ b2,
                     float* __restrict__ attn) {
  __shared__ float hid[64], y2[32];
  int b = blockIdx.x, t = threadIdx.x;
  const float* pc = pooled + b * 64;
  float acc = b1[t];
  for (int c = 0; c < 64; ++c) acc += pc[c] * w1[c * 64 + t];
  hid[t] = fmaxf(acc, 0.f);
  __syncthreads();
  if (t < 32) {
    float a2 = b2[t];
    for (int h = 0; h < 64; ++h) a2 += hid[h] * w2[h * 32 + t];
    y2[t] = a2;
  }
  __syncthreads();
  if (t < 4) {
    const float* yy = y2 + t * 8;
    float mx = yy[0];
    for (int j = 1; j < 8; ++j) mx = fmaxf(mx, yy[j]);
    float e[8], se = 0.f;
    for (int j = 0; j < 8; ++j) { e[j] = __expf(yy[j] - mx); se += e[j]; }
    for (int j = 0; j < 8; ++j) attn[b * 32 + t * 8 + j] = e[j] / se;
  }
}

// ---------------- host orchestration ----------------
extern "C" void kernel_launch(void* const* d_in, const int* in_sizes, int n_in,
                              void* d_out, int out_size, void* d_ws, size_t ws_size,
                              hipStream_t stream) {
  (void)in_sizes; (void)n_in; (void)out_size;
  // leaf order (jax pytree, sorted dict keys):
  // 0 conv2_w | 1..8 feb_res (w1,w2)x4 | 9 feb_w0 | 10 pre_w |
  // 11+st*16: dil3_dw,dil3_pw,dil5_dw,dil5_pw,dil7_dw,dil7_pw,
  //           sep1_dw,sep1_pw,sep3_dw,sep3_pw,sep5_dw,sep5_pw,sep7_dw,sep7_pw,
  //           w_cp1, w_out3  (4 steps)
  // 75 oa.b1 | 76 oa.b2 | 77 oa.w1 | 78 oa.w2 | 79 x
  const float* conv2_w = (const float*)d_in[0];
  const float* feb_w0  = (const float*)d_in[9];
  const float* pre_w   = (const float*)d_in[10];
  const float* oa_b1   = (const float*)d_in[75];
  const float* oa_b2   = (const float*)d_in[76];
  const float* oa_w1   = (const float*)d_in[77];
  const float* oa_w2   = (const float*)d_in[78];
  const float* x       = (const float*)d_in[79];

  const size_t ACT = (size_t)B_ * C_ * HW_ * 4;       // 16 MiB
  char* ws = (char*)d_ws;
  size_t off = 0;
  auto alloc = [&](size_t bytes) -> void* {
    void* p = ws + off; off += (bytes + 255) & ~(size_t)255; return p;
  };
  float* A      = (float*)alloc(ACT);
  float* Bbuf   = (float*)alloc(ACT);
  float* Cbuf   = (float*)alloc(ACT);
  unsigned short* states = (unsigned short*)alloc((size_t)B_ * 512 * HW_ * 2); // 64 MiB
  float* M2     = (float*)states;                      // reuse: same 64 MiB
  float* out1   = (float*)alloc((size_t)B_ * 256 * HW_ * 4);                   // 64 MiB
  unsigned short* WpRes[8];
  for (int i = 0; i < 8; ++i) WpRes[i] = (unsigned short*)alloc((size_t)9 * 64 * 64 * 2);
  unsigned short* WpPre = (unsigned short*)alloc((size_t)64 * 64 * 2);
  unsigned short* WpPw[4][7];
  for (int s = 0; s < 4; ++s)
    for (int j = 0; j < 7; ++j) WpPw[s][j] = (unsigned short*)alloc((size_t)64 * 64 * 2);
  unsigned short* WpCp[4];
  for (int s = 0; s < 4; ++s) WpCp[s] = (unsigned short*)alloc((size_t)256 * 512 * 2);
  unsigned short* WpW2[4];
  for (int s = 0; s < 4; ++s) WpW2[s] = (unsigned short*)alloc((size_t)64 * 256 * 2);
  float* pooled = (float*)alloc(B_ * 64 * 4);
  float* attn   = (float*)alloc(B_ * 32 * 4);
  if (off > ws_size) return;

  static const int dwoff[7] = {6, 8, 10, 12, 0, 2, 4};
  static const int pwoff[7] = {7, 9, 11, 13, 1, 3, 5};
  static const int kk[7]    = {1, 3, 5, 7, 3, 5, 7};
  static const int dl[7]    = {1, 1, 1, 1, 2, 2, 2};

  const dim3 BLK(256);
  const dim3 EW(B_ * C_ * HW_ / 256);                  // elementwise grids
  auto gemm_grid = [](int Co) { return dim3((Co / 16) * (HW_ / 16) / 8, 1, B_); };

  // ---- pack all weights to bf16 ----
  for (int i = 0; i < 8; ++i)
    k_pack_conv<<<(9 * 4096 + 255) / 256, BLK, 0, stream>>>((const float*)d_in[1 + i],
                                                            WpRes[i], 64, 64, 9);
  k_pack_conv<<<(4096 + 255) / 256, BLK, 0, stream>>>(pre_w, WpPre, 64, 64, 1);
  for (int s = 0; s < 4; ++s) {
    int base = 11 + s * 16;
    for (int j = 0; j < 7; ++j)
      k_pack_conv<<<(4096 + 255) / 256, BLK, 0, stream>>>((const float*)d_in[base + pwoff[j]],
                                                          WpPw[s][j], 64, 64, 1);
    k_pack_cp<<<(256 * 512 + 255) / 256, BLK, 0, stream>>>((const float*)d_in[base + 14], WpCp[s]);
    k_pack_w2<<<(64 * 256 + 255) / 256, BLK, 0, stream>>>((const float*)d_in[base + 15], WpW2[s]);
  }

  // ---- FEB: conv3x3 3->64, then 4 residual blocks (WMMA) ----
  k_conv_in<<<EW, BLK, 0, stream>>>(x, feb_w0, A);
  for (int rb = 0; rb < 4; ++rb) {
    k_conv3x3_wmma<<<dim3(512, 1, B_), BLK, 0, stream>>>(A, WpRes[rb * 2 + 0], Cbuf, nullptr, 0);
    k_conv3x3_wmma<<<dim3(512, 1, B_), BLK, 0, stream>>>(Cbuf, WpRes[rb * 2 + 1], A, A, 1);
  }

  // ---- attention ----
  k_gap<<<dim3(64, B_), BLK, 0, stream>>>(A, pooled);
  k_oa<<<dim3(B_), dim3(64), 0, stream>>>(pooled, oa_w1, oa_b1, oa_w2, oa_b2, attn);

  // ---- group: pre 1x1 (relu-in), then 4 operation steps ----
  k_gemm_wmma<<<gemm_grid(64), BLK, 0, stream>>>(WpPre, A, nullptr, Bbuf, nullptr,
                                                 nullptr, 0, 64, 64, 1, 64, 0);
  float* s_cur = Bbuf;
  float* s_oth = A;
  for (int st = 0; st < 4; ++st) {
    int base = 11 + st * 16;
    k_relu<<<EW, BLK, 0, stream>>>(s_cur, s_oth);                   // xr = relu(s0)
    for (int j = 0; j < 7; ++j) {
      k_dwconv<<<EW, BLK, 0, stream>>>(s_oth, (const float*)d_in[base + dwoff[j]],
                                       Cbuf, kk[j], dl[j]);
      // pointwise 64->64, epilogue: * attn -> bf16 states slot j
      k_gemm_wmma<<<gemm_grid(64), BLK, 0, stream>>>(WpPw[st][j], Cbuf, nullptr,
                                                     nullptr, states, attn, st * 8 + j,
                                                     64, 64, 0, 512, j * 64);
    }
    k_avgpool_states<<<EW, BLK, 0, stream>>>(s_cur, states, attn, st * 8 + 7);
    // CP contraction: out1[256,px] = Wcp[256,512] x states  (TDM-staged B)
    k_cp_gemm_tdm<<<dim3(HW_ / 16, 2, B_), BLK, 0, stream>>>(WpCp[st], states, out1);
    // per-pixel M2 = M*M  (writes into states region, fp32)
    k_m2<<<dim3(HW_ / 16, B_), BLK, 0, stream>>>(out1, M2);
    // y[64,px] = W2[64,256] x M2
    k_gemm_wmma<<<gemm_grid(64), BLK, 0, stream>>>(WpW2[st], M2, nullptr,
                                                   Cbuf, nullptr, nullptr, 0,
                                                   256, 64, 0, 64, 0);
    // instance-norm + softsign + residual + relu -> s_next
    k_norm<<<dim3(64, B_), BLK, 0, stream>>>(Cbuf, s_cur, s_oth);
    float* tmp = s_cur; s_cur = s_oth; s_oth = tmp;
  }

  // ---- final conv 64->3 ----
  k_conv_out<<<(B_ * 3 * HW_ + 255) / 256, BLK, 0, stream>>>(s_cur, conv2_w, (float*)d_out);
}
